// SlotAttention_83150566851448
// MI455X (gfx1250) — matrix operand order repaired and checked by reading, not compiled, not run on hardware
//
#include <hip/hip_runtime.h>

// ------------------------------------------------------------------
// SlotAttention for MI455X (gfx1250, wave32, WMMA bf16 -> f32 accum)
// v5: GEMM B-tile staging via gfx1250 GLOBAL_LOAD_ASYNC_TO_LDS_B128
//     (ASYNCcnt-tracked DMA, no VGPR round-trip); fallback to
//     global->reg->ds_store when the builtin is absent. Fixed ping-pong
//     LDS tiles, register-double-buffered B fragments, move-free
//     A-fragment pipelining (K % 64 == 0).
// ------------------------------------------------------------------

typedef unsigned short u16;
typedef unsigned int   u32;
typedef __attribute__((ext_vector_type(4)))  u32    u32x4;
typedef __attribute__((ext_vector_type(16))) __bf16 v16bf;
typedef __attribute__((ext_vector_type(8)))  float  v8f;

#define B_     2048
#define NS_    16
#define N_     36
#define D_     768
#define FS_    640
#define G3_    1920
#define AFS_   128
#define LN_EPS 1e-5f
#define SCALE_ 0.03608439182435161f   // 768^-0.5

#define BROW_  40                     // LDS B-tile row pitch (halves), bank-friendly

#if defined(__has_builtin)
#  if __has_builtin(__builtin_amdgcn_global_load_async_to_lds_b128)
#    define HAVE_ASYNC_LDS 1
#  endif
#endif
#ifndef HAVE_ASYNC_LDS
#  define HAVE_ASYNC_LDS 0
#endif

// builtin parameter types (from clang diagnostic: AS1 printed as __device__)
typedef int v4i_ __attribute__((vector_size(16)));
typedef __attribute__((address_space(1))) v4i_* gasp_t;   // global int4*
typedef __attribute__((address_space(3))) v4i_* lasp_t;   // LDS int4*

__device__ __forceinline__ void wait_async0() {
#if defined(__has_builtin) && __has_builtin(__builtin_amdgcn_s_wait_asynccnt)
  __builtin_amdgcn_s_wait_asynccnt(0);
#else
  asm volatile("s_wait_asynccnt 0x0" ::: "memory");
#endif
}

union Frag { u32x4 u[2]; u16 us[16]; v16bf v; };

__device__ __forceinline__ v8f v8f_zero() {
  v8f z = {0.f,0.f,0.f,0.f,0.f,0.f,0.f,0.f}; return z;
}

__device__ __forceinline__ u16 f2bf(float f) {
  u32 u = __float_as_uint(f);
  u += 0x7FFFu + ((u >> 16) & 1u);   // round-to-nearest-even
  return (u16)(u >> 16);
}

__device__ __forceinline__ float wsum32(float x) {
  x += __shfl_xor(x, 1, 32); x += __shfl_xor(x, 2, 32);
  x += __shfl_xor(x, 4, 32); x += __shfl_xor(x, 8, 32);
  x += __shfl_xor(x, 16, 32);
  return x;
}
__device__ __forceinline__ float wsum16(float x) {
  x += __shfl_xor(x, 1, 32); x += __shfl_xor(x, 2, 32);
  x += __shfl_xor(x, 4, 32); x += __shfl_xor(x, 8, 32);
  return x;
}
__device__ __forceinline__ float wmax16(float x) {
  x = fmaxf(x, __shfl_xor(x, 1, 32)); x = fmaxf(x, __shfl_xor(x, 2, 32));
  x = fmaxf(x, __shfl_xor(x, 4, 32)); x = fmaxf(x, __shfl_xor(x, 8, 32));
  return x;
}

// ---------------- elementwise utilities ----------------

__global__ __launch_bounds__(256) void k_f32_to_bf16(const float* __restrict__ in,
                                                     u16* __restrict__ out, int n) {
  int i = blockIdx.x * 256 + threadIdx.x;
  if (i < n) out[i] = f2bf(in[i]);
}

__global__ __launch_bounds__(256) void k_zero_u16(u16* __restrict__ p, int n) {
  int i = blockIdx.x * 256 + threadIdx.x;
  if (i < n) p[i] = 0;
}

__global__ __launch_bounds__(256) void k_init_slots(const float* __restrict__ cand,
                                                    float* __restrict__ sc, int n) {
  int i = blockIdx.x * 256 + threadIdx.x;
  if (i < n) { int r = i / FS_, c = i - r * FS_; sc[i] = cand[(size_t)r * D_ + c]; }
}

__global__ __launch_bounds__(256) void k_finalize(const float* __restrict__ sc,
                                                  const float* __restrict__ cand,
                                                  float* __restrict__ out, int n) {
  int i = blockIdx.x * 256 + threadIdx.x;
  if (i < n) {
    int r = i / D_, c = i - r * D_;
    out[i] = (c < FS_) ? sc[(size_t)r * FS_ + c] : cand[i];
  }
}

// ---------------- LayerNorm kernels (1 wave per row) ----------------

__global__ __launch_bounds__(256) void k_ln_pano(const float* __restrict__ x,
                                                 const float* __restrict__ g,
                                                 const float* __restrict__ be,
                                                 u16* __restrict__ out) {
  int lane = threadIdx.x & 31;
  int row  = blockIdx.x * 8 + (threadIdx.x >> 5);   // 0..73727
  int b    = row / N_;
  const float* xr = x + (size_t)row * D_;
  u16* o = out + (size_t)row * D_;
  float v[24];
  #pragma unroll
  for (int e = 0; e < 24; ++e) v[e] = xr[lane + e * 32];
  if (b < B_ - AFS_) {
    float s = 0.f;
    #pragma unroll
    for (int e = 0; e < 24; ++e) s += v[e];
    float mean = wsum32(s) * (1.f / D_);
    float s2 = 0.f;
    #pragma unroll
    for (int e = 0; e < 24; ++e) { float d = v[e] - mean; s2 += d * d; }
    float inv = rsqrtf(wsum32(s2) * (1.f / D_) + LN_EPS);
    #pragma unroll
    for (int e = 0; e < 24; ++e) {
      int c = lane + e * 32;
      v[e] = (v[e] - mean) * inv * g[c] + be[c];
    }
  }
  #pragma unroll
  for (int e = 0; e < 24; ++e) o[lane + e * 32] = f2bf(v[e]);
}

__global__ __launch_bounds__(256) void k_ln_slots(const float* __restrict__ sc,
                                                  const float* __restrict__ cand,
                                                  const float* __restrict__ g,
                                                  const float* __restrict__ be,
                                                  float* __restrict__ hf,
                                                  u16* __restrict__ hb,
                                                  u16* __restrict__ sq) {
  int lane = threadIdx.x & 31;
  int row  = blockIdx.x * 8 + (threadIdx.x >> 5);   // 0..32767
  const float* xr = sc + (size_t)row * FS_;
  float v[20];
  #pragma unroll
  for (int e = 0; e < 20; ++e) v[e] = xr[lane + e * 32];
  float s = 0.f;
  #pragma unroll
  for (int e = 0; e < 20; ++e) s += v[e];
  float mean = wsum32(s) * (1.f / FS_);
  float s2 = 0.f;
  #pragma unroll
  for (int e = 0; e < 20; ++e) { float d = v[e] - mean; s2 += d * d; }
  float inv = rsqrtf(wsum32(s2) * (1.f / FS_) + LN_EPS);
  #pragma unroll
  for (int e = 0; e < 20; ++e) {
    int c = lane + e * 32;
    float y = (v[e] - mean) * inv * g[c] + be[c];
    hf[(size_t)row * FS_ + c] = y;
    u16 t = f2bf(y);
    hb[(size_t)row * FS_ + c] = t;
    sq[(size_t)row * D_ + c]  = t;
  }
  #pragma unroll
  for (int e = 0; e < 4; ++e) {                    // tail 640..767 from cand_feat
    int c = FS_ + lane + e * 32;
    sq[(size_t)row * D_ + c] = f2bf(cand[(size_t)row * D_ + c]);
  }
}

__global__ __launch_bounds__(256) void k_gates_ln(const float* __restrict__ gi,
                                                  const float* __restrict__ gh,
                                                  const float* __restrict__ hf,
                                                  const float* __restrict__ g,
                                                  const float* __restrict__ be,
                                                  float* __restrict__ hnew,
                                                  u16* __restrict__ pre) {
  int lane = threadIdx.x & 31;
  int row  = blockIdx.x * 8 + (threadIdx.x >> 5);
  const float* gir = gi + (size_t)row * G3_;
  const float* ghr = gh + (size_t)row * G3_;
  float v[20];
  #pragma unroll
  for (int e = 0; e < 20; ++e) {
    int c = lane + e * 32;
    float ir = gir[c], iz = gir[FS_ + c], ig = gir[2 * FS_ + c];
    float hr = ghr[c], hz = ghr[FS_ + c], hg = ghr[2 * FS_ + c];
    float h  = hf[(size_t)row * FS_ + c];
    float r  = 1.f / (1.f + __expf(-(ir + hr)));
    float z  = 1.f / (1.f + __expf(-(iz + hz)));
    float gg = tanhf(ig + r * hg);
    float hn = (1.f - z) * gg + z * h;
    hnew[(size_t)row * FS_ + c] = hn;
    v[e] = hn;
  }
  float s = 0.f;
  #pragma unroll
  for (int e = 0; e < 20; ++e) s += v[e];
  float mean = wsum32(s) * (1.f / FS_);
  float s2 = 0.f;
  #pragma unroll
  for (int e = 0; e < 20; ++e) { float d = v[e] - mean; s2 += d * d; }
  float inv = rsqrtf(wsum32(s2) * (1.f / FS_) + LN_EPS);
  #pragma unroll
  for (int e = 0; e < 20; ++e) {
    int c = lane + e * 32;
    pre[(size_t)row * FS_ + c] = f2bf((v[e] - mean) * inv * g[c] + be[c]);
  }
}

// ---------------- WMMA GEMM: C = A(MxK) @ W(NxK)^T + bias ----------------
// 256 threads = 8 waves. Block tile 128(M) x 128(N); wave tile 16x128
// (8 accumulators). B tiles (128x32 bf16) ping-pong in LDS buf0/buf1
// (K unrolled by 64, buffer roles compile-time fixed). B tile staging
// uses GLOBAL_LOAD_ASYNC_TO_LDS_B128 when available (no VGPR round-trip,
// tracked by ASYNCcnt); B fragments double-buffered in registers; A
// fragments reloaded after last use. Requires K % 64 == 0 (768 or 640).
// mode bits: 1=bf16 out, 2=ReLU, 4=add residual (f32), 8=scatter store to vT(b,d,64)

__global__ __launch_bounds__(256) void k_gemm(const u16* __restrict__ A, int lda,
                                              const u16* __restrict__ Bw, int ldb,
                                              const float* __restrict__ bias,
                                              void* __restrict__ C, int ldc,
                                              const float* __restrict__ R, int ldr,
                                              int K, int mode) {
  __shared__ u16 Bs[2][128 * BROW_];

  int tid   = threadIdx.x;
  int lane  = tid & 31;
  int wave  = tid >> 5;
  int lrow  = lane & 15;
  int h     = lane >> 4;
  int mbase = blockIdx.x * 128 + wave * 16;
  int nbase = blockIdx.y * 128;

  int c0 = tid, c1 = tid + 256;                // 2 chunks of 8 halves per thread
  const u16* bg0 = Bw + (size_t)(nbase + (c0 >> 2)) * ldb + (c0 & 3) * 8;
  const u16* bg1 = Bw + (size_t)(nbase + (c1 >> 2)) * ldb + (c1 & 3) * 8;
  u16* bs0 = &Bs[0][(c0 >> 2) * BROW_ + (c0 & 3) * 8];
  u16* bs1 = &Bs[0][(c1 >> 2) * BROW_ + (c1 & 3) * 8];

  const u16* ap = A + (size_t)(mbase + lrow) * lda + h * 8;

  v8f acc[8];
  #pragma unroll
  for (int s = 0; s < 8; ++s) acc[s] = v8f_zero();

  // compute one 32-wide K-step from LDS buffer `buf` with A fragment `af`;
  // B fragments double-buffered so ds_loads overlap the WMMAs
  auto compute = [&](int buf, const Frag& af) {
    const u16* base = &Bs[buf][lrow * BROW_ + h * 8];
    Frag bf[2];
    bf[0].u[0] = *(const u32x4*)(base);
    bf[0].u[1] = *(const u32x4*)(base + 16);
    #pragma unroll
    for (int s = 0; s < 8; ++s) {
      int cur = s & 1, nxt = cur ^ 1;
      if (s < 7) {
        const u16* bp = base + (s + 1) * 16 * BROW_;
        bf[nxt].u[0] = *(const u32x4*)(bp);
        bf[nxt].u[1] = *(const u32x4*)(bp + 16);
      }
      acc[s] = __builtin_amdgcn_wmma_f32_16x16x32_bf16(
          false, af.v, false, bf[cur].v, (short)0, acc[s], false, false);
    }
  };

#if HAVE_ASYNC_LDS
  // stage B tile for K-offset kk directly into LDS buffer `buf` via the
  // async-to-LDS DMA path (ASYNCcnt); no VGPR data staging needed
  auto stageB = [&](int buf, int kk) {
    __builtin_amdgcn_global_load_async_to_lds_b128(
        (gasp_t)(const void*)(bg0 + kk),
        (lasp_t)(void*)(bs0 + buf * (128 * BROW_)), 0, 0);
    __builtin_amdgcn_global_load_async_to_lds_b128(
        (gasp_t)(const void*)(bg1 + kk),
        (lasp_t)(void*)(bs1 + buf * (128 * BROW_)), 0, 0);
  };

  stageB(0, 0);
  Frag af0, af1;
  af0.u[0] = *(const u32x4*)(ap);      af0.u[1] = *(const u32x4*)(ap + 16);
  af1.u[0] = *(const u32x4*)(ap + 32); af1.u[1] = *(const u32x4*)(ap + 48);
  wait_async0();
  __syncthreads();

  for (int kk = 0; kk < K; kk += 64) {
    stageB(1, kk + 32);                  // DMA next tile while computing
    compute(0, af0);
    af0.u[0] = *(const u32x4*)(ap + kk + 64);   // reload after last use
    af0.u[1] = *(const u32x4*)(ap + kk + 80);
    wait_async0();
    __syncthreads();

    stageB(0, kk + 64);
    compute(1, af1);
    af1.u[0] = *(const u32x4*)(ap + kk + 96);
    af1.u[1] = *(const u32x4*)(ap + kk + 112);
    wait_async0();
    __syncthreads();
  }
#else
  // fallback: global -> VGPR -> ds_store staging
  u32x4 r0 = *(const u32x4*)(bg0);
  u32x4 r1 = *(const u32x4*)(bg1);
  Frag af0, af1;
  af0.u[0] = *(const u32x4*)(ap);      af0.u[1] = *(const u32x4*)(ap + 16);
  af1.u[0] = *(const u32x4*)(ap + 32); af1.u[1] = *(const u32x4*)(ap + 48);
  *(u32x4*)(bs0) = r0;
  *(u32x4*)(bs1) = r1;
  __syncthreads();

  for (int kk = 0; kk < K; kk += 64) {
    r0 = *(const u32x4*)(bg0 + kk + 32);
    r1 = *(const u32x4*)(bg1 + kk + 32);
    compute(0, af0);
    af0.u[0] = *(const u32x4*)(ap + kk + 64);
    af0.u[1] = *(const u32x4*)(ap + kk + 80);
    *(u32x4*)(bs0 + 128 * BROW_) = r0;
    *(u32x4*)(bs1 + 128 * BROW_) = r1;
    __syncthreads();

    r0 = *(const u32x4*)(bg0 + kk + 64);
    r1 = *(const u32x4*)(bg1 + kk + 64);
    compute(1, af1);
    af1.u[0] = *(const u32x4*)(ap + kk + 96);
    af1.u[1] = *(const u32x4*)(ap + kk + 112);
    *(u32x4*)(bs0) = r0;
    *(u32x4*)(bs1) = r1;
    __syncthreads();
  }
#endif

  #pragma unroll
  for (int s = 0; s < 8; ++s) {
    int n = nbase + s * 16 + lrow;
    float bval = bias ? bias[n] : 0.f;
    #pragma unroll
    for (int r = 0; r < 8; ++r) {
      int m = mbase + h * 8 + r;
      float val = acc[s][r] + bval;
      if (mode & 2) val = fmaxf(val, 0.f);
      if (mode & 4) val += R[(size_t)m * ldr + n];
      if (mode & 8) {                        // v^T scatter: m=(b,j), col n=d
        int bb = m / N_, j = m - bb * N_;
        ((u16*)C)[((size_t)bb * D_ + n) * 64 + j] = f2bf(val);
      } else if (mode & 1) {
        ((u16*)C)[(size_t)m * ldc + n] = f2bf(val);
      } else {
        ((float*)C)[(size_t)m * ldc + n] = val;
      }
    }
  }
}

// ---------------- attention (1 wave per batch, all-register) ----------------
// dots^T[j,i] = sum_d k[j,d]*q[i,d]; softmax over i = 16-lane shfl reduction;
// accumulator layout == A-fragment layout for updates = attn @ v (via vT).
// Both loops register double-buffered (prefetch next fragments before WMMA).

__global__ __launch_bounds__(256) void k_attn(const u16* __restrict__ Q,
                                              const u16* __restrict__ Kb,
                                              const u16* __restrict__ VT,
                                              const unsigned char* __restrict__ mask,
                                              u16* __restrict__ U) {
  int lane = threadIdx.x & 31;
  int b    = blockIdx.x * 8 + (threadIdx.x >> 5);
  int lrow = lane & 15;
  int h    = lane >> 4;

  const u16* kb = Kb + (size_t)b * N_ * D_;
  const u16* qr = Q  + (size_t)(b * NS_ + lrow) * D_;   // B-frag column i = lrow

  auto load_dots = [&](int kk, Frag& fq, Frag& f0, Frag& f1, Frag& f2) {
    fq.u[0] = *(const u32x4*)(qr + kk + h * 8);
    fq.u[1] = *(const u32x4*)(qr + kk + h * 8 + 16);
    const u16* p0 = kb + (size_t)lrow * D_ + kk + h * 8;
    f0.u[0] = *(const u32x4*)(p0); f0.u[1] = *(const u32x4*)(p0 + 16);
    const u16* p1 = kb + (size_t)(16 + lrow) * D_ + kk + h * 8;
    f1.u[0] = *(const u32x4*)(p1); f1.u[1] = *(const u32x4*)(p1 + 16);
    const u16* p2 = kb + (size_t)(32 + lrow) * D_ + kk + h * 8;
    f2.u[0] = *(const u32x4*)(p2); f2.u[1] = *(const u32x4*)(p2 + 16);
  };

  v8f dt[3];
  dt[0] = v8f_zero(); dt[1] = v8f_zero(); dt[2] = v8f_zero();

  Frag bq[2], ak0[2], ak1[2], ak2[2];
  load_dots(0, bq[0], ak0[0], ak1[0], ak2[0]);
  #pragma unroll 2
  for (int i = 0; i < 24; ++i) {                  // 24 K-steps of 32
    int cur = i & 1, nxt = cur ^ 1;
    if (i < 23) load_dots(32 * (i + 1), bq[nxt], ak0[nxt], ak1[nxt], ak2[nxt]);
    dt[0] = __builtin_amdgcn_wmma_f32_16x16x32_bf16(
        false, ak0[cur].v, false, bq[cur].v, (short)0, dt[0], false, false);
    dt[1] = __builtin_amdgcn_wmma_f32_16x16x32_bf16(
        false, ak1[cur].v, false, bq[cur].v, (short)0, dt[1], false, false);
    dt[2] = __builtin_amdgcn_wmma_f32_16x16x32_bf16(
        false, ak2[cur].v, false, bq[cur].v, (short)0, dt[2], false, false);
  }

  // softmax over slot axis i (= lane%16) per column j; mask whole slot rows
  bool mi = mask[b * NS_ + lrow] != 0;
  float attn[24];
  #pragma unroll
  for (int t = 0; t < 3; ++t) {
    #pragma unroll
    for (int r = 0; r < 8; ++r) {
      float x = dt[t][r] * SCALE_;
      if (mi) x = -3.0e38f;
      float mx = wmax16(x);
      float e  = __expf(x - mx);
      float s  = wsum16(e);
      attn[t * 8 + r] = e / s;
    }
  }

  // pack attn into A-fragments for K-steps j in [0,32) and [32,64)
  Frag a0, a1;
  #pragma unroll
  for (int e = 0; e < 8; ++e) {
    a0.us[e]     = f2bf(attn[e]);                 // j = h*8 + e
    a0.us[8 + e] = f2bf(attn[8 + e]);             // j = 16 + h*8 + e
    int j2 = 32 + h * 8 + e;
    a1.us[e]     = (j2 < N_) ? f2bf(attn[16 + e]) : (u16)0;
    a1.us[8 + e] = 0;                             // j = 48..63 padding
  }

  // updates[i,d] = sum_j attn[i,j] * v[j,d], B-frags from vT(b, d, 64)
  const u16* vtb = VT + (size_t)b * D_ * 64;
  size_t ub = (size_t)b * NS_ * D_;

  auto load_v = [&](int nb, Frag& f0, Frag& f1) {
    const u16* vp = vtb + (size_t)(nb + lrow) * 64;
    f0.u[0] = *(const u32x4*)(vp + h * 8);
    f0.u[1] = *(const u32x4*)(vp + h * 8 + 16);
    f1.u[0] = *(const u32x4*)(vp + 32 + h * 8);
    f1.u[1] = *(const u32x4*)(vp + 32 + h * 8 + 16);
  };

  Frag vb0[2], vb1[2];
  load_v(0, vb0[0], vb1[0]);
  #pragma unroll 2
  for (int i = 0; i < 48; ++i) {                  // 48 N-tiles of 16
    int cur = i & 1, nxt = cur ^ 1;
    if (i < 47) load_v(16 * (i + 1), vb0[nxt], vb1[nxt]);
    v8f uacc = v8f_zero();
    uacc = __builtin_amdgcn_wmma_f32_16x16x32_bf16(false, a0.v, false, vb0[cur].v,
                                                   (short)0, uacc, false, false);
    uacc = __builtin_amdgcn_wmma_f32_16x16x32_bf16(false, a1.v, false, vb1[cur].v,
                                                   (short)0, uacc, false, false);
    int nb = 16 * i;
    #pragma unroll
    for (int r = 0; r < 8; ++r)
      U[ub + (size_t)(h * 8 + r) * D_ + nb + lrow] = f2bf(uacc[r]);
  }
}

// ------------------------------------------------------------------
// host orchestration
// ------------------------------------------------------------------

extern "C" void kernel_launch(void* const* d_in, const int* in_sizes, int n_in,
                              void* d_out, int out_size, void* d_ws, size_t ws_size,
                              hipStream_t stream) {
  const float* cand = (const float*)d_in[0];
  const float* pano = (const float*)d_in[1];
  const unsigned char* cmask = (const unsigned char*)d_in[2];   // jnp bool = 1 byte
  const float* Wq_ = (const float*)d_in[3];  const float* bq_ = (const float*)d_in[4];
  const float* Wk_ = (const float*)d_in[5];  const float* bk_ = (const float*)d_in[6];
  const float* Wv_ = (const float*)d_in[7];  const float* bv_ = (const float*)d_in[8];
  const float* gin = (const float*)d_in[9];  const float* bin = (const float*)d_in[10];
  const float* gsl = (const float*)d_in[11]; const float* bsl = (const float*)d_in[12];
  const float* gff = (const float*)d_in[13]; const float* bff = (const float*)d_in[14];
  const float* Wih_= (const float*)d_in[15]; const float* bih_= (const float*)d_in[16];
  const float* Whh_= (const float*)d_in[17]; const float* bhh_= (const float*)d_in[18];
  const float* W1_ = (const float*)d_in[19]; const float* b1_ = (const float*)d_in[20];
  const float* W2_ = (const float*)d_in[21]; const float* b2_ = (const float*)d_in[22];

  // ---- workspace arena ----
  char* W = (char*)d_ws;
  size_t off = 0;
  auto A_ = [&](size_t bytes) -> char* {
    char* p = W + off;
    off = (off + bytes + 255) & ~(size_t)255;
    return p;
  };
  u16* wq  = (u16*)A_((size_t)589824 * 2);
  u16* wk  = (u16*)A_((size_t)589824 * 2);
  u16* wv  = (u16*)A_((size_t)589824 * 2);
  u16* wih = (u16*)A_((size_t)1474560 * 2);
  u16* whh = (u16*)A_((size_t)1228800 * 2);
  u16* w1  = (u16*)A_((size_t)491520 * 2);
  u16* w2  = (u16*)A_((size_t)491520 * 2);
  u16* kbf = (u16*)A_((size_t)(73728 + 16) * D_ * 2);       // +16 pad rows for j tiles
  u16* vT  = (u16*)A_((size_t)B_ * D_ * 64 * 2);            // (b, d, j64)
  float* sc   = (float*)A_((size_t)32768 * FS_ * 4);        // slots core (state)
  float* hf   = (float*)A_((size_t)32768 * FS_ * 4);        // h = LN(slots) f32
  u16* hb_pre = (u16*)A_((size_t)32768 * FS_ * 2);          // h bf16, reused as `pre`
  u16* sq_act = (u16*)A_((size_t)32768 * D_ * 2);           // slotq bf16, reused as mlp act
  u16* qbf    = (u16*)A_((size_t)32768 * D_ * 2);
  u16* ubf    = (u16*)A_((size_t)32768 * D_ * 2);           // updates bf16
  float* gi   = (float*)A_((size_t)32768 * G3_ * 4);        // also aliases pano_bf16
  float* gh   = (float*)A_((size_t)32768 * G3_ * 4);
  float* hnew = (float*)A_((size_t)32768 * FS_ * 4);
  u16* pano_bf = (u16*)gi;                                  // pre-loop alias

  // ---- weight conversion to bf16 ----
  k_f32_to_bf16<<<2304, 256, 0, stream>>>(Wq_,  wq,  589824);
  k_f32_to_bf16<<<2304, 256, 0, stream>>>(Wk_,  wk,  589824);
  k_f32_to_bf16<<<2304, 256, 0, stream>>>(Wv_,  wv,  589824);
  k_f32_to_bf16<<<5760, 256, 0, stream>>>(Wih_, wih, 1474560);
  k_f32_to_bf16<<<4800, 256, 0, stream>>>(Whh_, whh, 1228800);
  k_f32_to_bf16<<<1920, 256, 0, stream>>>(W1_,  w1,  491520);
  k_f32_to_bf16<<<1920, 256, 0, stream>>>(W2_,  w2,  491520);

  // ---- zero k padding rows and vT (padding j>=36 must be 0) ----
  k_zero_u16<<<48, 256, 0, stream>>>(kbf + (size_t)73728 * D_, 16 * D_);
  k_zero_u16<<<393216, 256, 0, stream>>>(vT, 100663296);

  // ---- pano LayerNorm (first B-AFS batches) -> bf16 ----
  k_ln_pano<<<9216, 256, 0, stream>>>(pano, gin, bin, pano_bf);

  // ---- k / v projections (v stored transposed per-batch) ----
  dim3 gkv(576, 6);
  k_gemm<<<gkv, 256, 0, stream>>>(pano_bf, D_, wk, D_, bk_, kbf, D_, nullptr, 0, D_, 1);
  k_gemm<<<gkv, 256, 0, stream>>>(pano_bf, D_, wv, D_, bv_, vT,  0,  nullptr, 0, D_, 8);

  // ---- init slot state ----
  k_init_slots<<<81920, 256, 0, stream>>>(cand, sc, 32768 * FS_);

  // ---- slot-attention iterations ----
  for (int it = 0; it < 3; ++it) {
    k_ln_slots<<<4096, 256, 0, stream>>>(sc, cand, gsl, bsl, hf, hb_pre, sq_act);
    k_gemm<<<dim3(256, 6), 256, 0, stream>>>(sq_act, D_, wq, D_, bq_, qbf, D_,
                                             nullptr, 0, D_, 1);
    k_attn<<<256, 256, 0, stream>>>(qbf, kbf, vT, cmask, ubf);
    k_gemm<<<dim3(256, 15), 256, 0, stream>>>(ubf, D_, wih, D_, bih_, gi, G3_,
                                              nullptr, 0, D_, 0);
    k_gemm<<<dim3(256, 15), 256, 0, stream>>>(hb_pre, FS_, whh, FS_, bhh_, gh, G3_,
                                              nullptr, 0, FS_, 0);
    k_gates_ln<<<4096, 256, 0, stream>>>(gi, gh, hf, gff, bff, hnew, hb_pre);
    k_gemm<<<dim3(256, 6), 256, 0, stream>>>(hb_pre, FS_, w1, FS_, b1_, sq_act, D_,
                                             nullptr, 0, FS_, 1 | 2);
    k_gemm<<<dim3(256, 5), 256, 0, stream>>>(sq_act, D_, w2, D_, b2_, sc, FS_,
                                             hnew, FS_, D_, 4);
  }

  // ---- final output: [slots_core | cand tail] ----
  k_finalize<<<98304, 256, 0, stream>>>(sc, cand, (float*)d_out, 25165824);
}
